// SelfAttentive_52286931861928
// MI455X (gfx1250) — matrix operand order repaired and checked by reading, not compiled, not run on hardware
//
#include <hip/hip_runtime.h>
#include <hip/hip_bf16.h>
#include <math.h>

#define DIN   768
#define DH    256
#define NC    4
#define SLEN  2048
#define BATCH 64

#define BQ      128               // rows of X per workgroup in k1
#define APAD    8                 // bf16 padding (bank-conflict avoidance)
#define ASTRIDE (DIN + APAD)      // 776 (row stride, 1552 B = 16B aligned)
#define BSTRIDE (DIN + APAD)      // transposed-B row stride

typedef __attribute__((ext_vector_type(16))) __bf16 v16bf;
typedef __attribute__((ext_vector_type(8)))  __bf16 v8bf;   // 16 B, ds_load_b128
typedef __attribute__((ext_vector_type(8)))  float  v8f;

// Branch-free tanh with hardware exp: tanh(x) = 1 - 2/(e^{2x}+1).
// e^{2x}->inf gives 1, ->0 gives -1; no divergent libm range split.
__device__ __forceinline__ float fast_tanh(float x) {
    float e = __expf(2.0f * x);
    return 1.0f - __fdividef(2.0f, e + 1.0f);
}

__device__ __forceinline__ v16bf cat16(v8bf lo, v8bf hi) {
    return __builtin_shufflevector(lo, hi, 0, 1, 2, 3, 4, 5, 6, 7,
                                           8, 9, 10, 11, 12, 13, 14, 15);
}

// ---------------------------------------------------------------------------
// Kernel 1: logits[b,s,c] = tanh(X[b,s,:]@W1 + b1) @ ctx   (fused, X read once)
// 8 waves/block; wave w owns rows [blk*128 + 16w, +16). 16 N-tiles x 24 K-steps
// of v_wmma_f32_16x16x32_bf16; fragments loaded as 2x ds_load_b128 each.
// ---------------------------------------------------------------------------
__global__ __launch_bounds__(256)
void k1_logits(const float* __restrict__ X, const float* __restrict__ W1,
               const float* __restrict__ b1, const float* __restrict__ ctx,
               float* __restrict__ logits) {
    extern __shared__ char smem[];
    __bf16* ldsA  = (__bf16*)smem;                         // [BQ][ASTRIDE]
    __bf16* ldsBt = (__bf16*)(smem + BQ * ASTRIDE * 2);    // [16][BSTRIDE] (n-major)

    const int t     = threadIdx.x;
    const int wave  = t >> 5;
    const int L     = t & 31;
    const int lhalf = L >> 4;       // 0: lanes 0-15, 1: lanes 16-31
    const int l16   = L & 15;
    const long long row0 = (long long)blockIdx.x * BQ;     // global flattened row

    // Stage A block (fp32 -> bf16), coalesced: consecutive t -> consecutive k
    for (int e = t; e < BQ * DIN; e += 256) {
        int r = e / DIN;
        int k = e - r * DIN;
        ldsA[r * ASTRIDE + k] = (__bf16)X[(row0 + r) * DIN + k];
    }

    float plog[8][4];
#pragma unroll
    for (int r = 0; r < 8; ++r)
#pragma unroll
        for (int c = 0; c < 4; ++c) plog[r][c] = 0.0f;

    const int mrow = wave * 16 + l16;   // LDS row this lane gathers for A-frag
    // A-frag: j=0..7 -> K = kc + 8*lhalf + j ; j=8..15 -> +16. Two 16 B chunks.
    const __bf16* aBase0 = ldsA + mrow * ASTRIDE + (lhalf << 3);
    // B-frag (transposed LDS): N = l16 row, same K order as A-frag. Two 16 B chunks.
    const __bf16* bBase0 = ldsBt + l16 * BSTRIDE + (lhalf << 3);

    for (int n = 0; n < 16; ++n) {
        __syncthreads();   // previous B panel fully consumed / A staged
        // Stage W1 panel transposed: ldsBt[nn][k] = W1[k][n*16+nn]
        for (int e = t; e < DIN * 16; e += 256) {
            int k  = e >> 4;
            int nn = e & 15;
            ldsBt[nn * BSTRIDE + k] = (__bf16)W1[k * DH + n * 16 + nn];
        }
        __syncthreads();

        v8f acc = {};
        for (int kc = 0; kc < DIN; kc += 32) {
            v8bf alo = *(const v8bf*)(aBase0 + kc);        // K: kc+8*lhalf ..+7
            v8bf ahi = *(const v8bf*)(aBase0 + kc + 16);   // K: +16
            v8bf blo = *(const v8bf*)(bBase0 + kc);        // K: kc+8*lhalf ..+7
            v8bf bhi = *(const v8bf*)(bBase0 + kc + 16);   // K: +16
            acc = __builtin_amdgcn_wmma_f32_16x16x32_bf16(
                false, cat16(alo, ahi), false, cat16(blo, bhi),
                (short)0, acc, false, false);
        }

        // Epilogue: tanh(+bias), dot with the 4 context vectors
        const int col = n * 16 + l16;
        const float bias = b1[col];
        const float c0 = ctx[col * 4 + 0], c1 = ctx[col * 4 + 1];
        const float c2 = ctx[col * 4 + 2], c3 = ctx[col * 4 + 3];
#pragma unroll
        for (int r = 0; r < 8; ++r) {
            float h = fast_tanh(acc[r] + bias);
            plog[r][0] += h * c0; plog[r][1] += h * c1;
            plog[r][2] += h * c2; plog[r][3] += h * c3;
        }
    }

    // Reduce the 16 column-lanes of each half-wave (rows differ between halves)
#pragma unroll
    for (int r = 0; r < 8; ++r)
#pragma unroll
        for (int c = 0; c < 4; ++c) {
            float v = plog[r][c];
#pragma unroll
            for (int st = 1; st < 16; st <<= 1) v += __shfl_xor(v, st, 16);
            plog[r][c] = v;
        }

    if (l16 == 0) {
        long long rbase = row0 + wave * 16 + lhalf * 8;   // C/D: M = r + 8*lhalf
#pragma unroll
        for (int r = 0; r < 8; ++r)
#pragma unroll
            for (int c = 0; c < 4; ++c)
                logits[(rbase + r) * 4 + c] = plog[r][c];
    }
}

// ---------------------------------------------------------------------------
// Kernel 2: softmax over s (in place on the 2 MB logits buffer). 1 block per b.
// ---------------------------------------------------------------------------
__global__ __launch_bounds__(256)
void k2_softmax(float* __restrict__ lg) {
    const int b = blockIdx.x;
    const int t = threadIdx.x;
    __shared__ float red[256];
    float* base = lg + (long long)b * SLEN * NC;

    for (int c = 0; c < NC; ++c) {
        float m = -3.402823466e38f;
        for (int s = t; s < SLEN; s += 256) m = fmaxf(m, base[s * NC + c]);
        red[t] = m; __syncthreads();
        for (int st = 128; st > 0; st >>= 1) {
            if (t < st) red[t] = fmaxf(red[t], red[t + st]);
            __syncthreads();
        }
        m = red[0]; __syncthreads();

        float sum = 0.0f;
        for (int s = t; s < SLEN; s += 256) sum += __expf(base[s * NC + c] - m);
        red[t] = sum; __syncthreads();
        for (int st = 128; st > 0; st >>= 1) {
            if (t < st) red[t] += red[t + st];
            __syncthreads();
        }
        float inv = 1.0f / red[0]; __syncthreads();

        for (int s = t; s < SLEN; s += 256)
            base[s * NC + c] = __expf(base[s * NC + c] - m) * inv;
        __syncthreads();
    }
}

// ---------------------------------------------------------------------------
// Kernel 3: pooled[b, c*768+d] = relu(sum_s scores[b,s,c] * X[b,s,d])
// Second (final) streaming pass over X. grid (3, 64), 256 thr -> d = 0..767.
// ---------------------------------------------------------------------------
__global__ __launch_bounds__(256)
void k3_pool(const float* __restrict__ X, const float* __restrict__ scores,
             float* __restrict__ pooled) {
    const int b = blockIdx.y;
    const int d = blockIdx.x * 256 + threadIdx.x;
    const float*  xb = X + (long long)b * SLEN * DIN + d;
    const float4* sc = (const float4*)(scores + (long long)b * SLEN * NC);

    float a0 = 0.f, a1 = 0.f, a2 = 0.f, a3 = 0.f;
    for (int s = 0; s < SLEN; ++s) {
        // keep the HBM stream ahead of the FMAs (global_prefetch_b8)
        if (s + 32 < SLEN) __builtin_prefetch(xb + (long long)(s + 32) * DIN, 0, 1);
        float  x = xb[(long long)s * DIN];
        float4 w = sc[s];
        a0 += w.x * x; a1 += w.y * x; a2 += w.z * x; a3 += w.w * x;
    }
    float* pb = pooled + (long long)b * (NC * DIN);
    pb[0 * DIN + d] = fmaxf(a0, 0.f);
    pb[1 * DIN + d] = fmaxf(a1, 0.f);
    pb[2 * DIN + d] = fmaxf(a2, 0.f);
    pb[3 * DIN + d] = fmaxf(a3, 0.f);
}

// ---------------------------------------------------------------------------
// Kernel 4: out = pooled[64,3072] @ W2[3072,768] + b2, WMMA bf16.
// One wave per 16x16 output tile: grid 4*48 = 192 blocks of 32 threads.
// pooled (0.75 MB) and W2 (9.4 MB) are L2-resident.
// ---------------------------------------------------------------------------
__global__ __launch_bounds__(32)
void k4_out(const float* __restrict__ pooled, const float* __restrict__ W2,
            const float* __restrict__ b2, float* __restrict__ out) {
    const int mt = blockIdx.x & 3;
    const int nt = blockIdx.x >> 2;
    const int L  = threadIdx.x;
    const int lhalf = L >> 4, l16 = L & 15;
    const int m0 = mt * 16, n0 = nt * 16;

    v8f acc = {};
    for (int kc = 0; kc < NC * DIN; kc += 32) {
        v16bf av, bv;
#pragma unroll
        for (int j = 0; j < 16; ++j) {
            int ka = kc + ((j >> 3) << 4) + (lhalf << 3) + (j & 7);
            av[j] = (__bf16)pooled[(m0 + l16) * (NC * DIN) + ka];
            int kb = kc + ((j >> 3) << 4) + (lhalf << 3) + (j & 7);
            bv[j] = (__bf16)W2[(long long)kb * DIN + n0 + l16];
        }
        acc = __builtin_amdgcn_wmma_f32_16x16x32_bf16(
            false, av, false, bv, (short)0, acc, false, false);
    }

    const float bias = b2[n0 + l16];
#pragma unroll
    for (int r = 0; r < 8; ++r)
        out[(long long)(m0 + r + (lhalf << 3)) * DIN + n0 + l16] = acc[r] + bias;
}

// ---------------------------------------------------------------------------
extern "C" void kernel_launch(void* const* d_in, const int* in_sizes, int n_in,
                              void* d_out, int out_size, void* d_ws, size_t ws_size,
                              hipStream_t stream) {
    const float* X   = (const float*)d_in[0];   // [64, 2048, 768]
    const float* W1  = (const float*)d_in[1];   // [768, 256]
    const float* b1  = (const float*)d_in[2];   // [256]
    const float* ctx = (const float*)d_in[3];   // [256, 4]
    const float* W2  = (const float*)d_in[4];   // [3072, 768]
    const float* b2  = (const float*)d_in[5];   // [768]
    float* out = (float*)d_out;                 // [64, 768]

    float* logits = (float*)d_ws;                                   // 2 MB
    float* pooled = (float*)((char*)d_ws +
                             (size_t)BATCH * SLEN * NC * sizeof(float)); // 0.75 MB

    const size_t lds1 = (size_t)BQ * ASTRIDE * 2 + (size_t)16 * BSTRIDE * 2; // ~219 KB

    k1_logits<<<(BATCH * SLEN) / BQ, 256, lds1, stream>>>(X, W1, b1, ctx, logits);
    k2_softmax<<<BATCH, 256, 0, stream>>>(logits);
    k3_pool<<<dim3(3, BATCH), 256, 0, stream>>>(X, logits, pooled);
    k4_out<<<4 * (DIN / 16), 32, 0, stream>>>(pooled, W2, b2, out);
}
